// C2Former_67216238183264
// MI455X (gfx1250) — compile-verified, matrix operand order checked by reading, not run on hardware
//
#include <hip/hip_runtime.h>

// ---------------------------------------------------------------------------
// C2Former for MI455X (gfx1250): all GEMM-like work (conv1x1, conv3x3 via
// im2col, attention QK^T and A*V) runs on v_wmma_f32_16x16x32_f16 (f16 in,
// f32 accumulate). Elementwise / irregular ops (dwconv5, LN+GELU offset net,
// bilinear grid-sample, instance-norm stats, softmax) are fp32 VALU kernels.
// ---------------------------------------------------------------------------

typedef _Float16 h16;
typedef __attribute__((ext_vector_type(16))) _Float16 v16h;
typedef __attribute__((ext_vector_type(8)))  _Float16 v8h;
typedef __attribute__((ext_vector_type(8)))  float    v8f;

#define CB   2        // batch
#define CC   256      // channels
#define CH   48
#define CHW  2304     // 48*48
#define LDSP 40       // padded LDS K-stride (32 + 8 halves; rows 16B-aligned)

// ---------------------------------------------------------------------------
// Generic batched GEMM: D[M,N] = A[M,K] (f16, row-major) x B[K,N] (f16,
// row-major) + bias.  256 threads = 8 waves; block tile 64x64; each wave owns
// a 16(M) x 32(N) piece (two 16x16 WMMA accumulators).  K multiple of 32,
// N multiple of 64 (always 2304 here).  All global pointers / row strides are
// 16B-aligned -> unconditional v8h (b128) loads.
// EPI: 0 = none, 1 = relu.   GM: 1 = M-bounds guard needed (only A*V, M=32).
// ---------------------------------------------------------------------------
template<int EPI, int GM>
__global__ __launch_bounds__(256) void gemm_wmma_f16(
    const h16* __restrict__ A, long sA,
    const h16* __restrict__ B, long sB,
    const float* __restrict__ bias,
    float* __restrict__ out32, h16* __restrict__ out16, long sC,
    int M, int N, int K)
{
    __shared__ h16 As[64 * LDSP];
    __shared__ h16 Bs[64 * LDSP];
    const int tid  = threadIdx.x;
    const int lane = tid & 31, wave = tid >> 5;
    const int wm = wave >> 1, wn = wave & 1;       // 4 waves in M, 2 in N
    const int bm = blockIdx.y * 64, bn = blockIdx.x * 64;
    A += (long)blockIdx.z * sA;
    B += (long)blockIdx.z * sB;

    const int hf = lane >> 4;      // which 16-lane half
    const int mr = lane & 15;      // row (A) / col (B) index within fragment

    // staging coordinates (uniform per thread across k-steps)
    const int arow = tid >> 2, acol = (tid & 3) * 8;   // A: 64 rows x 32 cols
    const int bkk  = tid >> 3, bnn  = (tid & 7) * 8;   // B: 32 k    x 64 n

    v8f acc0 = {}; v8f acc1 = {};

    for (int k0 = 0; k0 < K; k0 += 32) {
        // stage A tile 64x32 (row-major), one b128 load + one b128 LDS store
        {
            v8h vv = {};
            if (!GM || bm + arow < M)
                vv = *(const v8h*)(A + (long)(bm + arow) * K + k0 + acol);
            *(v8h*)&As[arow * LDSP + acol] = vv;
        }
        // stage B tile 32x64 transposed -> Bs[n][k]; one b128 load, 8 b16 stores
        {
            v8h vv = *(const v8h*)(B + (long)(k0 + bkk) * N + bn + bnn);
            #pragma unroll
            for (int j = 0; j < 8; ++j) Bs[(bnn + j) * LDSP + bkk] = vv[j];
        }
        if (k0 + 32 < K) {          // prefetch next K-tile (global_prefetch_b8)
            if (!GM || bm + arow < M)
                __builtin_prefetch(A + (long)(bm + arow) * K + k0 + 32 + acol, 0, 1);
            __builtin_prefetch(B + (long)(k0 + 32 + bkk) * N + bn + bnn, 0, 1);
        }
        __syncthreads();

        // fragments per ISA 16-bit layout: K = (v/4)*16 + half*8 + (v&3)*2
        v16h a = {}; v16h b0 = {}; v16h b1 = {};
        #pragma unroll
        for (int v = 0; v < 8; ++v) {
            int kl = (v >> 2) * 16 + hf * 8 + (v & 3) * 2;
            a [2*v]   = As[(wm*16 + mr) * LDSP + kl];
            a [2*v+1] = As[(wm*16 + mr) * LDSP + kl + 1];
            b0[2*v]   = Bs[(wn*32 + mr) * LDSP + kl];
            b0[2*v+1] = Bs[(wn*32 + mr) * LDSP + kl + 1];
            b1[2*v]   = Bs[(wn*32 + 16 + mr) * LDSP + kl];
            b1[2*v+1] = Bs[(wn*32 + 16 + mr) * LDSP + kl + 1];
        }
        acc0 = __builtin_amdgcn_wmma_f32_16x16x32_f16(false, a, false, b0,
                                                      (short)0, acc0, false, false);
        acc1 = __builtin_amdgcn_wmma_f32_16x16x32_f16(false, a, false, b1,
                                                      (short)0, acc1, false, false);
        __syncthreads();
    }

    // epilogue: C/D layout -> m = r + 8*half, n = lane&15
    #pragma unroll
    for (int r = 0; r < 8; ++r) {
        int m = bm + wm * 16 + hf * 8 + r;
        if (GM && m >= M) continue;
        float bv = bias ? bias[m] : 0.f;
        float v0 = acc0[r] + bv;
        float v1 = acc1[r] + bv;
        if (EPI == 1) { v0 = fmaxf(v0, 0.f); v1 = fmaxf(v1, 0.f); }
        int n0 = bn + wn * 32 + mr;
        long base = (long)blockIdx.z * sC + (long)m * N;
        if (out32) { out32[base + n0] = v0; out32[base + n0 + 16] = v1; }
        if (out16) { out16[base + n0] = (h16)v0; out16[base + n0 + 16] = (h16)v1; }
    }
}

// --------------------------- helper kernels --------------------------------

__global__ void cvt_f16_k(const float* __restrict__ x, h16* __restrict__ o, int n) {
    int i = blockIdx.x * blockDim.x + threadIdx.x;
    if (i < n) o[i] = (h16)x[i];
}

// xcat16[b][ch][p], ch<256 -> vis, else lwir
__global__ void concat_f16_k(const float* __restrict__ a, const float* __restrict__ b,
                             h16* __restrict__ o, int n) {
    int i = blockIdx.x * blockDim.x + threadIdx.x;
    if (i >= n) return;
    int p = i % CHW;
    int rest = i / CHW;
    int ch = rest % 512;
    int bb = rest / 512;
    float v = (ch < CC) ? a[((long)bb * CC + ch) * CHW + p]
                        : b[((long)bb * CC + (ch - CC)) * CHW + p];
    o[i] = (h16)v;
}

// depthwise 5x5, pad 2: in/out [8][64][48][48]
__global__ void dwconv5_k(const float* __restrict__ x, const float* __restrict__ w,
                          const float* __restrict__ bias, float* __restrict__ o, int n) {
    int i = blockIdx.x * blockDim.x + threadIdx.x;
    if (i >= n) return;
    int p = i % CHW;
    int c = (i / CHW) & 63;
    const float* xi = x + (long)(i / CHW) * CHW;
    const float* wc = w + c * 25;
    int py = p / CH, px = p % CH;
    float s = bias[c];
    #pragma unroll
    for (int ky = 0; ky < 5; ++ky) {
        int yy = py + ky - 2;
        if (yy < 0 || yy >= CH) continue;
        #pragma unroll
        for (int kx = 0; kx < 5; ++kx) {
            int xx = px + kx - 2;
            if (xx < 0 || xx >= CH) continue;
            s += wc[ky * 5 + kx] * xi[yy * CH + xx];
        }
    }
    o[i] = s;
}

// per-pixel channel LayerNorm(64) -> exact GELU -> 1x1 (2 ch) -> tanh*2/48
__global__ void offset_net_k(const float* __restrict__ off, const float* __restrict__ lnw,
                             const float* __restrict__ lnb, const float* __restrict__ wpw,
                             float* __restrict__ offs, int n) {
    int i = blockIdx.x * blockDim.x + threadIdx.x;   // bg*CHW + p
    if (i >= n) return;
    int p = i % CHW, bg = i / CHW;
    const float* src = off + (long)bg * 64 * CHW + p;
    float s = 0.f, sq = 0.f;
    for (int c = 0; c < 64; ++c) { float v = src[c * CHW]; s += v; sq += v * v; }
    float mu  = s / 64.f;
    float var = sq / 64.f - mu * mu;
    float inv = rsqrtf(var + 1e-5f);
    float o0 = 0.f, o1 = 0.f;
    for (int c = 0; c < 64; ++c) {
        float v = (src[c * CHW] - mu) * inv * lnw[c] + lnb[c];
        float g = 0.5f * v * (1.f + erff(v * 0.70710678118654752f));
        o0 += wpw[c] * g;
        o1 += wpw[64 + c] * g;
    }
    float fac = 2.0f / 48.0f;                        // FACTOR * (1/Hk)
    offs[i * 2]     = tanhf(o0) * fac;               // (y, x) order
    offs[i * 2 + 1] = tanhf(o1) * fac;
}

// bilinear grid_sample, align_corners=True, zero padding; out f16 [B][C][HW]
__global__ void sample_k(const float* __restrict__ x, const float* __restrict__ offs,
                         h16* __restrict__ o, int use_off, int n) {
    int i = blockIdx.x * blockDim.x + threadIdx.x;   // (bg*64 + c)*CHW + p
    if (i >= n) return;
    int p = i % CHW;
    int bgc = i / CHW;
    int bg = bgc >> 6;
    int py = p / CH, px = p % CH;
    float ry = ((py + 0.5f) / 48.f) * 2.f - 1.f;
    float rx = ((px + 0.5f) / 48.f) * 2.f - 1.f;
    if (use_off) {
        ry += offs[((long)bg * CHW + p) * 2];
        rx += offs[((long)bg * CHW + p) * 2 + 1];
    }
    float gx = (rx + 1.f) * 0.5f * 47.f;
    float gy = (ry + 1.f) * 0.5f * 47.f;
    float x0f = floorf(gx), y0f = floorf(gy);
    float wx1 = gx - x0f, wy1 = gy - y0f;
    int x0 = (int)x0f, y0 = (int)y0f;
    const float* img = x + (long)bgc * CHW;
    float acc = 0.f;
    #pragma unroll
    for (int dy = 0; dy < 2; ++dy)
        #pragma unroll
        for (int dx = 0; dx < 2; ++dx) {
            int xi = x0 + dx, yi = y0 + dy;
            float w = (dx ? wx1 : 1.f - wx1) * (dy ? wy1 : 1.f - wy1);
            if (xi >= 0 && xi < CH && yi >= 0 && yi < CH)
                acc += w * img[yi * CH + xi];
        }
    o[i] = (h16)acc;
}

// per-(b,c) sum & sumsq over HW
__global__ void stats_k(const float* __restrict__ x, float* __restrict__ mean,
                        float* __restrict__ sumsq) {
    __shared__ float s1[256], s2[256];
    int bc = blockIdx.x;
    const float* src = x + (long)bc * CHW;
    float a = 0.f, b = 0.f;
    for (int i = threadIdx.x; i < CHW; i += 256) { float v = src[i]; a += v; b += v * v; }
    s1[threadIdx.x] = a; s2[threadIdx.x] = b;
    __syncthreads();
    for (int off = 128; off > 0; off >>= 1) {
        if (threadIdx.x < off) {
            s1[threadIdx.x] += s1[threadIdx.x + off];
            s2[threadIdx.x] += s2[threadIdx.x + off];
        }
        __syncthreads();
    }
    if (threadIdx.x == 0) { mean[bc] = s1[0] / (float)CHW; sumsq[bc] = s2[0]; }
}

// im2col for 3x3 pad1: out16[b][(ci*9+s)][p], K-order matches (C,C,3,3) weights
__global__ void im2col3_k(const float* __restrict__ x, h16* __restrict__ o, int n) {
    int i = blockIdx.x * blockDim.x + threadIdx.x;
    if (i >= n) return;
    int p = i % CHW;
    int rest = i / CHW;
    int s = rest % 9;
    int bc = rest / 9;                       // b*256 + ci
    int py = p / CH, px = p % CH;
    int yy = py + s / 3 - 1, xx = px + s % 3 - 1;
    float v = 0.f;
    if (yy >= 0 && yy < CH && xx >= 0 && xx < CH) v = x[(long)bc * CHW + yy * CH + xx];
    o[i] = (h16)v;
}

// fused modality-norm combine: ref_n*(gamma+lr_std) + (beta+lr_mean) -> f16
__global__ void mn_combine_k(const float* __restrict__ ref, const float* __restrict__ gamma,
                             const float* __restrict__ beta,
                             const float* __restrict__ meanR, const float* __restrict__ m2R,
                             const float* __restrict__ meanL, const float* __restrict__ m2L,
                             h16* __restrict__ o, int n) {
    int i = blockIdx.x * blockDim.x + threadIdx.x;
    if (i >= n) return;
    int bc = i / CHW;
    float muR  = meanR[bc];
    float varR = m2R[bc] / (float)CHW - muR * muR;   // biased (InstanceNorm)
    float inv  = rsqrtf(varR + 1e-5f);
    float muL  = meanL[bc];
    float vs   = (m2L[bc] - (float)CHW * muL * muL) / (float)(CHW - 1); // unbiased
    float stdL = sqrtf(fmaxf(vs, 0.f));
    float rn = (ref[i] - muR) * inv;
    o[i] = (h16)(rn * (gamma[i] + stdL) + (beta[i] + muL));
}

// q f32 [hb*32 + c][m] -> qT f16 [hb][m][c]   (A operand for QK^T)
__global__ void transpose_q_k(const float* __restrict__ q, h16* __restrict__ qt, int n) {
    int i = blockIdx.x * blockDim.x + threadIdx.x;   // hb*CHW*32 + m*32 + c
    if (i >= n) return;
    int c = i & 31;
    int m = (i >> 5) % CHW;
    int hb = i / (CHW * 32);
    qt[i] = (h16)q[((long)hb * 32 + c) * CHW + m];
}

// row softmax of S[m][n]*scale, output transposed f16 PT[n][m]
__global__ void softmax_t_k(const float* __restrict__ S, h16* __restrict__ PT, float scale) {
    __shared__ float red[256];
    int m = blockIdx.x;
    const float* row = S + (long)m * CHW;
    float mx = -1e30f;
    for (int i = threadIdx.x; i < CHW; i += 256) mx = fmaxf(mx, row[i] * scale);
    red[threadIdx.x] = mx; __syncthreads();
    for (int off = 128; off > 0; off >>= 1) {
        if (threadIdx.x < off) red[threadIdx.x] = fmaxf(red[threadIdx.x], red[threadIdx.x + off]);
        __syncthreads();
    }
    mx = red[0]; __syncthreads();
    float sum = 0.f;
    for (int i = threadIdx.x; i < CHW; i += 256) sum += expf(row[i] * scale - mx);
    red[threadIdx.x] = sum; __syncthreads();
    for (int off = 128; off > 0; off >>= 1) {
        if (threadIdx.x < off) red[threadIdx.x] += red[threadIdx.x + off];
        __syncthreads();
    }
    float inv = 1.f / red[0];
    for (int i = threadIdx.x; i < CHW; i += 256)
        PT[(long)i * CHW + m] = (h16)(expf(row[i] * scale - mx) * inv);
}

// ------------------------------- launcher ----------------------------------

extern "C" void kernel_launch(void* const* d_in, const int* in_sizes, int n_in,
                              void* d_out, int out_size, void* d_ws, size_t ws_size,
                              hipStream_t stream) {
    (void)in_sizes; (void)n_in; (void)out_size; (void)ws_size;

    const float* vis_x    = (const float*)d_in[0];
    const float* lwir_x   = (const float*)d_in[1];
    const float* w_combinq= (const float*)d_in[2];
    const float* b_combinq= (const float*)d_in[3];
    const float* w_off_dw = (const float*)d_in[4];
    const float* b_off_dw = (const float*)d_in[5];
    const float* ln_w     = (const float*)d_in[6];
    const float* ln_b     = (const float*)d_in[7];
    const float* w_off_pw = (const float*)d_in[8];
    // vismn weights 9..14, lwirmn 15..20, attention w/b pairs 21..36

    char* wsp = (char*)d_ws;
    auto alloc = [&](size_t bytes) { void* p = (void*)wsp;
                                     wsp += (bytes + 255) & ~(size_t)255; return p; };
    const size_t NBC = (size_t)CB * CC * CHW;        // 1,179,648

    h16*   xcat16  = (h16*)  alloc((size_t)CB * 512 * CHW * 2);
    float* cq      = (float*)alloc(NBC * 4);
    float* offdw   = (float*)alloc(NBC * 4);
    float* offs    = (float*)alloc((size_t)8 * CHW * 2 * 4);
    h16*   viss16  = (h16*)  alloc(NBC * 2);
    h16*   lwirs16 = (h16*)  alloc(NBC * 2);
    h16*   ic16    = (h16*)  alloc((size_t)CB * 2304 * CHW * 2);
    float* hbuf    = (float*)alloc(NBC * 4);
    float* gbuf    = (float*)alloc(NBC * 4);
    float* bbuf    = (float*)alloc(NBC * 4);
    float* meanR   = (float*)alloc(512 * 4);
    float* m2R     = (float*)alloc(512 * 4);
    float* meanL   = (float*)alloc(512 * 4);
    float* m2L     = (float*)alloc(512 * 4);
    h16*   mn16    = (h16*)  alloc(NBC * 2);
    float* qbuf    = (float*)alloc(NBC * 4);
    h16*   qT_lwir = (h16*)  alloc(NBC * 2);
    h16*   qT_vis  = (h16*)  alloc(NBC * 2);
    h16*   kvis16  = (h16*)  alloc(NBC * 2);
    h16*   vvis16  = (h16*)  alloc(NBC * 2);
    h16*   klwir16 = (h16*)  alloc(NBC * 2);
    h16*   vlwir16 = (h16*)  alloc(NBC * 2);
    float* Sbuf    = (float*)alloc((size_t)CHW * CHW * 4);
    h16*   PT16    = (h16*)  alloc((size_t)CHW * CHW * 2);
    h16*   attn16  = (h16*)  alloc(NBC * 2);
    h16*   wcq16   = (h16*)  alloc((size_t)256 * 512 * 2);
    h16*   w3x3[6];
    for (int j = 0; j < 6; ++j) w3x3[j] = (h16*)alloc((size_t)256 * 2304 * 2);
    h16*   w1x1[8];
    for (int j = 0; j < 8; ++j) w1x1[j] = (h16*)alloc((size_t)256 * 256 * 2);

    auto blocks = [](long n) { return dim3((unsigned)((n + 255) / 256)); };

    // guard: 1 only for the A*V GEMM (M=32, not a multiple of the 64 tile)
    auto gemm = [&](int epi, int guard, const h16* Aw, long sA, const h16* Bx, long sB,
                    const float* bias, float* o32, h16* o16, long sC,
                    int M, int N, int K, int batches) {
        dim3 grid((unsigned)(N / 64), (unsigned)((M + 63) / 64), (unsigned)batches);
        if (guard)
            gemm_wmma_f16<0,1><<<grid, 256, 0, stream>>>(Aw, sA, Bx, sB, bias, o32, o16, sC, M, N, K);
        else if (epi)
            gemm_wmma_f16<1,0><<<grid, 256, 0, stream>>>(Aw, sA, Bx, sB, bias, o32, o16, sC, M, N, K);
        else
            gemm_wmma_f16<0,0><<<grid, 256, 0, stream>>>(Aw, sA, Bx, sB, bias, o32, o16, sC, M, N, K);
    };

    // --- weight conversion to f16 ---
    cvt_f16_k<<<blocks(256 * 512), 256, 0, stream>>>(w_combinq, wcq16, 256 * 512);
    const int w3idx[6] = {9, 11, 13, 15, 17, 19};
    for (int j = 0; j < 6; ++j)
        cvt_f16_k<<<blocks(256 * 2304), 256, 0, stream>>>((const float*)d_in[w3idx[j]],
                                                          w3x3[j], 256 * 2304);
    const int w1idx[8] = {21, 23, 25, 27, 29, 31, 33, 35};
    for (int j = 0; j < 8; ++j)
        cvt_f16_k<<<blocks(256 * 256), 256, 0, stream>>>((const float*)d_in[w1idx[j]],
                                                         w1x1[j], 256 * 256);

    // --- combin_q = conv1x1(concat(vis,lwir)) ---
    {
        int n = CB * 512 * CHW;
        concat_f16_k<<<blocks(n), 256, 0, stream>>>(vis_x, lwir_x, xcat16, n);
        gemm(0, 0, wcq16, 0, xcat16, (long)512 * CHW, b_combinq, cq, nullptr,
             (long)CC * CHW, CC, CHW, 512, CB);
    }
    // --- offset net ---
    {
        int n = 8 * 64 * CHW;
        dwconv5_k<<<blocks(n), 256, 0, stream>>>(cq, w_off_dw, b_off_dw, offdw, n);
        int n2 = 8 * CHW;
        offset_net_k<<<blocks(n2), 256, 0, stream>>>(offdw, ln_w, ln_b, w_off_pw, offs, n2);
        sample_k<<<blocks(n), 256, 0, stream>>>(vis_x,  offs, viss16,  1, n);
        sample_k<<<blocks(n), 256, 0, stream>>>(lwir_x, offs, lwirs16, 0, n);
    }

    // --- modality norm + q projection (run twice) ---
    auto run_mn = [&](const float* lr, const float* ref,
                      const h16* wc, const float* bc, const h16* wg, const float* bg_,
                      const h16* wb, const float* bb_,
                      const h16* wq, const float* bq, h16* qT) {
        stats_k<<<dim3(CB * CC), 256, 0, stream>>>(ref, meanR, m2R);
        stats_k<<<dim3(CB * CC), 256, 0, stream>>>(lr,  meanL, m2L);
        long nic = (long)CB * CC * 9 * CHW;
        im2col3_k<<<blocks(nic), 256, 0, stream>>>(lr, ic16, (int)nic);
        gemm(1, 0, wc, 0, ic16, (long)2304 * CHW, bc, hbuf, nullptr,
             (long)CC * CHW, CC, CHW, 2304, CB);                     // h = relu(conv3x3)
        im2col3_k<<<blocks(nic), 256, 0, stream>>>(hbuf, ic16, (int)nic);
        gemm(0, 0, wg, 0, ic16, (long)2304 * CHW, bg_, gbuf, nullptr,
             (long)CC * CHW, CC, CHW, 2304, CB);                     // gamma
        gemm(0, 0, wb, 0, ic16, (long)2304 * CHW, bb_, bbuf, nullptr,
             (long)CC * CHW, CC, CHW, 2304, CB);                     // beta
        int nel = (int)NBC;
        mn_combine_k<<<blocks(nel), 256, 0, stream>>>(ref, gbuf, bbuf, meanR, m2R,
                                                      meanL, m2L, mn16, nel);
        gemm(0, 0, wq, 0, mn16, (long)CC * CHW, bq, qbuf, nullptr,
             (long)CC * CHW, CC, CHW, CC, CB);                       // q projection
        transpose_q_k<<<blocks(nel), 256, 0, stream>>>(qbuf, qT, nel);
    };
    // q_lwir: lr=vis_x, ref=lwir_x, vismn weights
    run_mn(vis_x, lwir_x, w3x3[0], (const float*)d_in[10], w3x3[1], (const float*)d_in[12],
           w3x3[2], (const float*)d_in[14], w1x1[0], (const float*)d_in[22], qT_lwir);
    // q_vis: lr=lwir_x, ref=vis_x, lwirmn weights
    run_mn(lwir_x, vis_x, w3x3[3], (const float*)d_in[16], w3x3[4], (const float*)d_in[18],
           w3x3[5], (const float*)d_in[20], w1x1[1], (const float*)d_in[24], qT_vis);

    // --- k/v projections on sampled features ---
    gemm(0, 0, w1x1[2], 0, viss16,  (long)CC * CHW, (const float*)d_in[26], nullptr, kvis16,
         (long)CC * CHW, CC, CHW, CC, CB);
    gemm(0, 0, w1x1[3], 0, viss16,  (long)CC * CHW, (const float*)d_in[28], nullptr, vvis16,
         (long)CC * CHW, CC, CHW, CC, CB);
    gemm(0, 0, w1x1[4], 0, lwirs16, (long)CC * CHW, (const float*)d_in[30], nullptr, klwir16,
         (long)CC * CHW, CC, CHW, CC, CB);
    gemm(0, 0, w1x1[5], 0, lwirs16, (long)CC * CHW, (const float*)d_in[32], nullptr, vlwir16,
         (long)CC * CHW, CC, CHW, CC, CB);

    // --- attention: loop 16 head-batches, reuse S / PT scratch ---
    const float SCALE = 0.17677669529663687f;        // 32^-0.5
    auto attend = [&](const h16* qT, const h16* kmat, const h16* vmat) {
        for (int hb = 0; hb < 16; ++hb) {
            // S[m][n] = sum_c qT[m][c] * k[c][n]  (single K=32 WMMA step)
            gemm(0, 0, qT + (size_t)hb * CHW * 32, 0, kmat + (size_t)hb * 32 * CHW, 0,
                 nullptr, Sbuf, nullptr, 0, CHW, CHW, 32, 1);
            softmax_t_k<<<dim3(CHW), 256, 0, stream>>>(Sbuf, PT16, SCALE);
            // O[c][m] = sum_n V[c][n] * PT[n][m]   (M=32 -> guarded tile)
            gemm(0, 1, vmat + (size_t)hb * 32 * CHW, 0, PT16, 0,
                 nullptr, nullptr, attn16 + (size_t)hb * 32 * CHW, 0, 32, CHW, CHW, 1);
        }
    };

    attend(qT_lwir, kvis16, vvis16);
    gemm(0, 0, w1x1[6], 0, attn16, (long)CC * CHW, (const float*)d_in[34],
         (float*)d_out, nullptr, (long)CC * CHW, CC, CHW, CC, CB);   // out_vis

    attend(qT_vis, klwir16, vlwir16);
    gemm(0, 0, w1x1[7], 0, attn16, (long)CC * CHW, (const float*)d_in[36],
         (float*)d_out + NBC, nullptr, (long)CC * CHW, CC, CHW, CC, CB); // out_lwir
}
